// FocusOnSpark_90838558311174
// MI455X (gfx1250) — matrix operand order repaired and checked by reading, not compiled, not run on hardware
//
#include <hip/hip_runtime.h>

// ---------------- sizes (match reference) ----------------
#define Bn 8
#define Nn 16384
#define Fk 1024
#define Dd 512
#define Ss 64
#define Cc 4
#define Hh 8
#define DH 64
#define LM 1024
#define Mm 512            // LM/2
#define SIM_TH 0.7f
#define GAMMA  0.8f

// ---------------- WMMA types / helpers ----------------
typedef __attribute__((ext_vector_type(16))) __bf16 v16bf;
typedef __attribute__((ext_vector_type(8)))  float  v8f;

union ABf {
  v16bf v;
  unsigned short u[16];
  uint4 q4[2];
};

__device__ __forceinline__ unsigned short f2bf(float f) {
  unsigned u = __float_as_uint(f);
  unsigned r = u + 0x7FFFu + ((u >> 16) & 1u);   // RNE
  return (unsigned short)(r >> 16);
}
__device__ __forceinline__ float bf2f(unsigned bits) {
  return __uint_as_float(bits << 16);
}

#define WMMA_BF16(a, b, c) \
  __builtin_amdgcn_wmma_f32_16x16x32_bf16(false, (a), false, (b), (short)0, (c), false, false)

// A fragment: 16x32 bf16, row-major source with leading dim lda (elements).
// Lane l: row m=l&15, chunks at K = (l>>4)*8 and 16+(l>>4)*8 (ISA 05_wmma 16-bit A layout).
__device__ __forceinline__ v16bf load_a_frag(const unsigned short* A, int lda,
                                             size_t row0, int k0, int lane) {
  int m  = lane & 15;
  int kb = (lane >> 4) * 8;
  const unsigned short* r = A + (row0 + (size_t)m) * (size_t)lda + k0;
  ABf f;
  f.q4[0] = *(const uint4*)(r + kb);
  f.q4[1] = *(const uint4*)(r + 16 + kb);
  return f.v;
}

// B fragment: 32x16 bf16 (KxN) sourced from an N-major ("transposed") buffer Bt[n][k].
// Lane l: col n=l&15, chunks at K = (l>>4)*8 and 16+(l>>4)*8 (matches A layout).
__device__ __forceinline__ v16bf load_b_frag(const unsigned short* Bt, int ldb,
                                             size_t n0, int k0, int lane) {
  int n  = lane & 15;
  int kb = (lane >> 4) * 8;
  const unsigned short* r = Bt + (n0 + (size_t)n) * (size_t)ldb + k0 + kb;
  ABf f;
  f.q4[0] = *(const uint4*)(r);
  f.q4[1] = *(const uint4*)(r + 16);
  return f.v;
}

__device__ __forceinline__ unsigned mono_key(float f) {
  unsigned u = __float_as_uint(f);
  return (u & 0x80000000u) ? ~u : (u | 0x80000000u);
}

// ---------------- init ----------------
__global__ void init_kernel(int* prune, int* nkept) {
  int i = blockIdx.x * 256 + threadIdx.x;
  if (i < Bn * Nn) prune[i] = 0;
  if (i < Bn) nkept[i] = 0;
}

// ---------------- text surrogate: masked mean pool + projection ----------------
__global__ __launch_bounds__(256) void text_kernel(const int* ids, const int* am,
                                                   const float* emb, const float* Wtp,
                                                   const float* btp, float* text_emb) {
  __shared__ float tf[768];
  __shared__ float cnt_s;
  int b = blockIdx.x, t = threadIdx.x;
  float a0 = 0.f, a1 = 0.f, a2 = 0.f, cnt = 0.f;
  for (int s = 0; s < Ss; ++s) {
    int m = am[b * Ss + s];
    if (t == 0) cnt += (float)m;
    if (m) {
      const float* row = emb + (size_t)ids[b * Ss + s] * 768;
      a0 += row[t]; a1 += row[t + 256]; a2 += row[t + 512];
    }
  }
  if (t == 0) cnt_s = fmaxf(cnt, 1.f);
  __syncthreads();
  float inv = 1.f / cnt_s;
  tf[t] = a0 * inv; tf[t + 256] = a1 * inv; tf[t + 512] = a2 * inv;
  __syncthreads();
  for (int d = t; d < Dd; d += 256) {
    float s = btp[d];
    for (int e = 0; e < 768; ++e) s += tf[e] * Wtp[(size_t)e * Dd + d];
    text_emb[b * Dd + d] = s;
  }
}

// ---------------- weight transpose + bf16 convert: Wt[n][k] = bf16(W[k][n]) ----------------
__global__ void wt_kernel(const float* W, unsigned short* Wt, int K, int Ncol) {
  int lin = blockIdx.x * 256 + threadIdx.x;
  if (lin >= K * Ncol) return;
  int n = lin / K, k = lin - n * K;
  Wt[lin] = f2bf(W[(size_t)k * Ncol + n]);
}

// ---------------- fused: h = bags@W_enc + b ; LayerNorm ; ReLU ; row-norm ; bf16 store -----
// block: 256 threads (8 waves), 16 rows x 512 cols; wave w -> col tiles 4w..4w+3.
// A tile double-buffered in LDS (one barrier per K step); B fragments stream from L2.
__global__ __launch_bounds__(256) void enc_kernel(const float* bags, const float* benc,
                                                  const float* lng, const float* lnb,
                                                  const unsigned short* WtEnc,
                                                  unsigned short* img, float* inv_norm) {
  __shared__ __align__(16) unsigned short Alds[2][16][40];
  __shared__ float hlds[16][Dd];
  __shared__ float red[16][17];
  __shared__ float red2[16][17];
  __shared__ float mu_s[16], rstd_s[16];

  int t = threadIdx.x, lane = t & 31, w = t >> 5;
  size_t row0 = (size_t)blockIdx.x * 16;
  const float* Arow = bags + row0 * Fk;

  v8f z = {0.f, 0.f, 0.f, 0.f, 0.f, 0.f, 0.f, 0.f};
  v8f acc[4] = {z, z, z, z};

  int sr = t >> 4;                // staging row 0..15
  int sc = (t & 15) * 2;          // staging col (even)
  {                               // prologue: stage K-step 0 into buffer 0
    float2 v2 = *(const float2*)(Arow + (size_t)sr * Fk + sc);
    Alds[0][sr][sc] = f2bf(v2.x);
    Alds[0][sr][sc + 1] = f2bf(v2.y);
  }
  for (int it = 0; it < Fk / 32; ++it) {
    int cur = it & 1;
    int k0 = it * 32;
    __syncthreads();              // staged buffer visible; prior-iter reads drained
    v16bf a = load_a_frag(&Alds[cur][0][0], 40, 0, 0, lane);
    if (it + 1 < Fk / 32) {       // stage next K-step into the other buffer
      const float* src = Arow + (size_t)sr * Fk + k0 + 32 + sc;
      float2 v2 = *(const float2*)src;
      Alds[cur ^ 1][sr][sc] = f2bf(v2.x);
      Alds[cur ^ 1][sr][sc + 1] = f2bf(v2.y);
      __builtin_prefetch((const void*)(src + 32), 0, 3);  // WGP-scope prefetch
    }
#pragma unroll
    for (int ct = 0; ct < 4; ++ct) {
      v16bf bf = load_b_frag(WtEnc, Fk, (size_t)(w * 64 + ct * 16), k0, lane);
      acc[ct] = WMMA_BF16(a, bf, acc[ct]);
    }
  }

  // spill C tiles (+bias) into LDS for cross-wave LayerNorm
  int mh = 8 * (lane >> 4);
#pragma unroll
  for (int ct = 0; ct < 4; ++ct) {
    int col = w * 64 + ct * 16 + (lane & 15);
    float bias = benc[col];
#pragma unroll
    for (int j = 0; j < 8; ++j) hlds[j + mh][col] = acc[ct][j] + bias;
  }
  __syncthreads();

  int r = t >> 4, cl = t & 15;    // 16 threads per row, 32 contiguous cols each
  float s = 0.f, s2 = 0.f;
  for (int j = 0; j < 32; ++j) {
    float v = hlds[r][cl * 32 + j];
    s += v; s2 += v * v;
  }
  red[r][cl] = s; red2[r][cl] = s2;
  __syncthreads();
  if (t < 16) {
    float a = 0.f, b2 = 0.f;
    for (int j = 0; j < 16; ++j) { a += red[t][j]; b2 += red2[t][j]; }
    float mu = a / (float)Dd;
    float var = b2 / (float)Dd - mu * mu;
    mu_s[t] = mu; rstd_s[t] = rsqrtf(var + 1e-5f);
  }
  __syncthreads();
  float mu = mu_s[r], rs = rstd_s[r], ss = 0.f;
  unsigned short* orow = img + (row0 + (size_t)r) * Dd;
  for (int j = 0; j < 32; ++j) {
    int c = cl * 32 + j;
    float v = (hlds[r][c] - mu) * rs * lng[c] + lnb[c];
    v = fmaxf(v, 0.f);
    ss += v * v;
    orow[c] = f2bf(v);
  }
  red[r][cl] = ss;
  __syncthreads();
  if (t < 16) {
    float a = 0.f;
    for (int j = 0; j < 16; ++j) a += red[t][j];
    inv_norm[row0 + (size_t)t] = 1.f / (sqrtf(a) + 1e-8f);
  }
}

// ---------------- stage 1: banded cosine-similarity pruning via WMMA ----------------
// 128 threads = 4 waves; each wave owns one 16-row i-block, checks 3 predecessor j-blocks.
__global__ __launch_bounds__(128) void sim_kernel(const unsigned short* img,
                                                  const float* inv_norm,
                                                  const unsigned char* pad_mask,
                                                  int* prune) {
  int t = threadIdx.x, lane = t & 31, w = t >> 5;
  int iblk = blockIdx.x * 4 + w;             // 0 .. B*N/16-1
  int b = iblk >> 10;                        // 1024 i-blocks per batch
  int lblk = iblk & 1023;
  size_t base = (size_t)b * Nn;
  size_t i0 = base + (size_t)lblk * 16;
  v8f z = {0.f, 0.f, 0.f, 0.f, 0.f, 0.f, 0.f, 0.f};

  for (int jb = 0; jb < 3; ++jb) {
    int j0l = lblk * 16 - 32 + jb * 16;
    if (j0l < 0) continue;                    // wave-uniform
    size_t j0 = base + (size_t)j0l;
    v8f acc = z;
    for (int k0 = 0; k0 < Dd; k0 += 32) {
      v16bf a  = load_a_frag(img, Dd, i0, k0, lane);
      v16bf bf = load_b_frag(img, Dd, j0, k0, lane);   // B[k][n] = img[j0+n][k]
      acc = WMMA_BF16(a, bf, acc);
    }
    int jg = (int)j0 + (lane & 15);
    bool jvalid = (pad_mask[jg] == 0);
    float invj = inv_norm[jg];
    int mh = 8 * (lane >> 4);
#pragma unroll
    for (int j = 0; j < 8; ++j) {
      int ig = (int)i0 + j + mh;
      int d = ig - jg;
      if (d >= 1 && d <= 32 && jvalid) {
        float sim = acc[j] * inv_norm[ig] * invj;
        if (sim >= SIM_TH) atomicOr(&prune[ig], 1);
      }
    }
  }
}

// ---------------- stage 2a: rel = img . text_emb, masked; count keeps ----------------
__global__ __launch_bounds__(256) void rel_kernel(const unsigned short* img, const float* temb,
                                                  const unsigned char* pad_mask, const int* prune,
                                                  float* relm, int* nkept) {
  int t = threadIdx.x, lane = t & 31, w = t >> 5;
  size_t row = (size_t)blockIdx.x * 8 + w;
  int b = (int)(row >> 14);
  const uint4* ir4 = (const uint4*)(img + row * Dd);
  const float* te = temb + (size_t)b * Dd;
  float s = 0.f;
  for (int c = lane; c < Dd / 8; c += 32) {
    uint4 qv = ir4[c];
    const float* tp = te + c * 8;
    s += bf2f(qv.x & 0xFFFFu) * tp[0] + bf2f(qv.x >> 16) * tp[1];
    s += bf2f(qv.y & 0xFFFFu) * tp[2] + bf2f(qv.y >> 16) * tp[3];
    s += bf2f(qv.z & 0xFFFFu) * tp[4] + bf2f(qv.z >> 16) * tp[5];
    s += bf2f(qv.w & 0xFFFFu) * tp[6] + bf2f(qv.w >> 16) * tp[7];
  }
  for (int off = 16; off; off >>= 1) s += __shfl_xor(s, off, 32);
  if (lane == 0) {
    bool keep = (pad_mask[row] == 0) && (prune[row] == 0);
    relm[row] = keep ? s : -__builtin_inff();
    if (keep) atomicAdd(&nkept[b], 1);
  }
}

// ---------------- stage 2b: exact radix-select top-k_b + ascending-index compaction -------
__global__ __launch_bounds__(1024) void select_kernel(const float* relm, const int* nkept,
                                                      int* idx_out) {
  __shared__ unsigned hist[256];
  __shared__ unsigned scan[1024];
  __shared__ unsigned sh_prefix, sh_remaining, sh_cgt, sh_run;
  int b = blockIdx.x, t = threadIdx.x;
  const float* rb = relm + (size_t)b * Nn;

  int kb = (int)(GAMMA * (float)nkept[b]);
  kb = min(max(kb, 1), LM);

  if (t == 0) { sh_prefix = 0; sh_remaining = (unsigned)kb; sh_cgt = 0; }
  __syncthreads();

  for (int p = 0; p < 4; ++p) {
    int shift = 24 - 8 * p;
    if (t < 256) hist[t] = 0;
    __syncthreads();
    unsigned pref = sh_prefix;
    unsigned pmask = p ? (0xFFFFFFFFu << (shift + 8)) : 0u;
    for (int i = t; i < Nn; i += 1024) {
      unsigned key = mono_key(rb[i]);
      if ((key & pmask) == (pref & pmask)) atomicAdd(&hist[(key >> shift) & 0xFFu], 1u);
    }
    __syncthreads();
    if (t == 0) {
      unsigned c = 0, rem = sh_remaining, chosen = 0;
      for (int bi = 255; bi >= 0; --bi) {
        unsigned h = hist[bi];
        if (c + h >= rem) { chosen = (unsigned)bi; break; }
        c += h;
      }
      sh_cgt += c;
      sh_remaining = rem - c;
      sh_prefix = pref | (chosen << shift);
    }
    __syncthreads();
  }
  unsigned T = sh_prefix;        // exact key of k_b-th largest
  unsigned neq = sh_remaining;   // how many ==T to take (lowest indices first)

  if (t == 0) sh_run = 0;
  __syncthreads();
  int* ob = idx_out + b * LM;
  for (int c0 = 0; c0 < Nn; c0 += 1024) {
    int i = c0 + t;
    unsigned key = mono_key(rb[i]);
    unsigned gt = (key > T) ? 1u : 0u;
    unsigned eq = (key == T) ? 1u : 0u;
    unsigned v = (gt << 16) | eq;
    scan[t] = v;
    __syncthreads();
    for (int off = 1; off < 1024; off <<= 1) {
      unsigned addv = (t >= off) ? scan[t - off] : 0u;
      __syncthreads();
      scan[t] += addv;
      __syncthreads();
    }
    unsigned incl = scan[t];
    unsigned excl = incl - v;
    unsigned run = sh_run;
    unsigned gt_rank = (run >> 16) + (excl >> 16);
    unsigned eq_rank = (run & 0xFFFFu) + (excl & 0xFFFFu);
    bool sel = gt || (eq && eq_rank < neq);
    if (sel) ob[gt_rank + min(eq_rank, neq)] = i;    // ascending by construction
    __syncthreads();
    if (t == 1023) sh_run = run + incl;
    __syncthreads();
  }
  unsigned total = (sh_run >> 16) + min(sh_run & 0xFFFFu, neq);  // == k_b
  for (int p = (int)total + t; p < LM; p += 1024) ob[p] = Nn;    // sentinel pad
}

// ---------------- stage 3: masked adjacent-pair mean pooling -> feat3 (bf16) ----------------
__global__ __launch_bounds__(128) void pair_kernel(const unsigned short* img, const int* idx,
                                                   unsigned short* feat3) {
  int b = blockIdx.y, m = blockIdx.x, t = threadIdx.x;
  int i0 = idx[b * LM + 2 * m], i1 = idx[b * LM + 2 * m + 1];
  float m0 = (i0 < Nn) ? 1.f : 0.f, m1 = (i1 < Nn) ? 1.f : 0.f;
  const unsigned short* r0 = img + ((size_t)b * Nn + (size_t)min(i0, Nn - 1)) * Dd;
  const unsigned short* r1 = img + ((size_t)b * Nn + (size_t)min(i1, Nn - 1)) * Dd;
  float inv = 1.f / fmaxf(m0 + m1, 1.f);
  unsigned short* o = feat3 + ((size_t)b * Mm + (size_t)m) * Dd;
  for (int d = t; d < Dd; d += 128)
    o[d] = f2bf((bf2f(r0[d]) * m0 + bf2f(r1[d]) * m1) * inv);
}

// ---------------- K/V projections: C = feat3 @ W + bias (WMMA, f32 out) ----------------
__global__ __launch_bounds__(256) void kv_kernel(const unsigned short* A, const unsigned short* Wt,
                                                 const float* bias, float* Cout) {
  int t = threadIdx.x, lane = t & 31, w = t >> 5;
  size_t row0 = (size_t)blockIdx.y * Mm + (size_t)blockIdx.x * 16;
  v8f z = {0.f, 0.f, 0.f, 0.f, 0.f, 0.f, 0.f, 0.f};
  v8f acc[4] = {z, z, z, z};
  for (int k0 = 0; k0 < Dd; k0 += 32) {
    v16bf a = load_a_frag(A, Dd, row0, k0, lane);
#pragma unroll
    for (int ct = 0; ct < 4; ++ct) {
      v16bf bf = load_b_frag(Wt, Dd, (size_t)(w * 64 + ct * 16), k0, lane);
      acc[ct] = WMMA_BF16(a, bf, acc[ct]);
    }
  }
  int mh = 8 * (lane >> 4);
#pragma unroll
  for (int ct = 0; ct < 4; ++ct) {
    int col = w * 64 + ct * 16 + (lane & 15);
    float bi = bias[col];
#pragma unroll
    for (int j = 0; j < 8; ++j)
      Cout[(row0 + (size_t)(j + mh)) * Dd + col] = acc[ct][j] + bi;
  }
}

// ---------------- q projection (tiny) ----------------
__global__ __launch_bounds__(256) void q_kernel(const float* temb, const float* Wq,
                                                const float* bq, float* q) {
  int b = blockIdx.x, t = threadIdx.x;
  const float* te = temb + (size_t)b * Dd;
  for (int d = t; d < Dd; d += 256) {
    float s = bq[d];
    for (int e = 0; e < Dd; ++e) s += te[e] * Wq[(size_t)e * Dd + d];
    q[b * Dd + d] = s;
  }
}

// ---------------- cross-modal MHA (one query per batch, M=512 keys) ----------------
__global__ __launch_bounds__(256) void attn_kernel(const float* q, const float* kbuf,
                                                   const float* vbuf, float* ctx) {
  __shared__ float at[Mm];
  __shared__ float qs[DH];
  __shared__ float rmax[256];
  __shared__ float part[4][DH];
  int h = blockIdx.x, b = blockIdx.y, t = threadIdx.x;
  if (t < DH) qs[t] = q[b * Dd + h * DH + t];
  __syncthreads();
  float sc[2], loc = -__builtin_inff();
#pragma unroll
  for (int r = 0; r < 2; ++r) {
    int m = t * 2 + r;
    const float* kr = kbuf + ((size_t)b * Mm + m) * Dd + h * DH;
    float s = 0.f;
    for (int e = 0; e < DH; ++e) s += qs[e] * kr[e];
    sc[r] = s * 0.125f;            // 1/sqrt(64)
    loc = fmaxf(loc, sc[r]);
  }
  rmax[t] = loc; __syncthreads();
  for (int off = 128; off; off >>= 1) { if (t < off) rmax[t] = fmaxf(rmax[t], rmax[t + off]); __syncthreads(); }
  float mx = rmax[0];
  __syncthreads();
  float e0 = __expf(sc[0] - mx), e1 = __expf(sc[1] - mx);
  at[t * 2] = e0; at[t * 2 + 1] = e1;
  rmax[t] = e0 + e1; __syncthreads();
  for (int off = 128; off; off >>= 1) { if (t < off) rmax[t] += rmax[t + off]; __syncthreads(); }
  float invs = 1.f / rmax[0];
  int d = t & 63, ch = t >> 6;
  float a = 0.f;
  for (int m = ch * 128; m < ch * 128 + 128; ++m)
    a += at[m] * vbuf[((size_t)b * Mm + m) * Dd + h * DH + d];
  part[ch][d] = a; __syncthreads();
  if (t < DH)
    ctx[b * Dd + h * DH + t] = (part[0][t] + part[1][t] + part[2][t] + part[3][t]) * invs;
}

// ---------------- output head: ctx@Wo+bo then @W_cls+b_cls ----------------
__global__ __launch_bounds__(256) void out_kernel(const float* ctx, const float* Wo,
                                                  const float* bo, const float* Wcls,
                                                  const float* bcls, float* logits) {
  __shared__ float o[Dd];
  __shared__ float pc[Cc][64];
  int b = blockIdx.x, t = threadIdx.x;
  const float* cb = ctx + (size_t)b * Dd;
  for (int d = t; d < Dd; d += 256) {
    float s = bo[d];
    for (int e = 0; e < Dd; ++e) s += cb[e] * Wo[(size_t)e * Dd + d];
    o[d] = s;
  }
  __syncthreads();
  int c = t >> 6, l = t & 63;
  float s = 0.f;
  for (int d = l; d < Dd; d += 64) s += o[d] * Wcls[(size_t)d * Cc + c];
  pc[c][l] = s; __syncthreads();
  if (t < Cc) {
    float a = 0.f;
    for (int j = 0; j < 64; ++j) a += pc[t][j];
    logits[b * Cc + t] = a + bcls[t];
  }
}

// ---------------- host ----------------
extern "C" void kernel_launch(void* const* d_in, const int* in_sizes, int n_in,
                              void* d_out, int out_size, void* d_ws, size_t ws_size,
                              hipStream_t stream) {
  const float* bags          = (const float*)d_in[0];
  const unsigned char* pmask = (const unsigned char*)d_in[1];   // bool [B,N]
  const int* ids             = (const int*)d_in[2];
  const int* am              = (const int*)d_in[3];
  const float* emb           = (const float*)d_in[4];
  const float* Wtp           = (const float*)d_in[5];
  const float* btp           = (const float*)d_in[6];
  const float* Wenc          = (const float*)d_in[7];
  const float* benc          = (const float*)d_in[8];
  const float* lng           = (const float*)d_in[9];
  const float* lnb           = (const float*)d_in[10];
  const float* Wq            = (const float*)d_in[11];
  const float* bq            = (const float*)d_in[12];
  const float* Wk            = (const float*)d_in[13];
  const float* bk            = (const float*)d_in[14];
  const float* Wv            = (const float*)d_in[15];
  const float* bv            = (const float*)d_in[16];
  const float* Wo            = (const float*)d_in[17];
  const float* bo            = (const float*)d_in[18];
  const float* Wcls          = (const float*)d_in[19];
  const float* bcls          = (const float*)d_in[20];
  float* logits              = (float*)d_out;

  char* ws = (char*)d_ws;
  size_t off = 0;
  auto alloc = [&](size_t bytes) -> void* {
    void* p = ws + off;
    off += (bytes + 255) & ~(size_t)255;
    return p;
  };
  unsigned short* img   = (unsigned short*)alloc((size_t)Bn * Nn * Dd * 2);
  float*          invn  = (float*)alloc((size_t)Bn * Nn * 4);
  int*            prune = (int*)alloc((size_t)Bn * Nn * 4);
  float*          relm  = (float*)alloc((size_t)Bn * Nn * 4);
  int*            nkept = (int*)alloc((size_t)Bn * 4);
  int*            idx   = (int*)alloc((size_t)Bn * LM * 4);
  float*          temb  = (float*)alloc((size_t)Bn * Dd * 4);
  unsigned short* WtEnc = (unsigned short*)alloc((size_t)Fk * Dd * 2);
  unsigned short* Wkt   = (unsigned short*)alloc((size_t)Dd * Dd * 2);
  unsigned short* Wvt   = (unsigned short*)alloc((size_t)Dd * Dd * 2);
  unsigned short* feat3 = (unsigned short*)alloc((size_t)Bn * Mm * Dd * 2);
  float*          kbuf  = (float*)alloc((size_t)Bn * Mm * Dd * 4);
  float*          vbuf  = (float*)alloc((size_t)Bn * Mm * Dd * 4);
  float*          qbuf  = (float*)alloc((size_t)Bn * Dd * 4);
  float*          ctx   = (float*)alloc((size_t)Bn * Dd * 4);
  if (off > ws_size) return;  // workspace too small: bail (no OOB writes)

  init_kernel<<<(Bn * Nn + 255) / 256, 256, 0, stream>>>(prune, nkept);
  text_kernel<<<Bn, 256, 0, stream>>>(ids, am, emb, Wtp, btp, temb);
  wt_kernel<<<(Fk * Dd + 255) / 256, 256, 0, stream>>>(Wenc, WtEnc, Fk, Dd);
  wt_kernel<<<(Dd * Dd + 255) / 256, 256, 0, stream>>>(Wk, Wkt, Dd, Dd);
  wt_kernel<<<(Dd * Dd + 255) / 256, 256, 0, stream>>>(Wv, Wvt, Dd, Dd);

  enc_kernel<<<(Bn * Nn) / 16, 256, 0, stream>>>(bags, benc, lng, lnb, WtEnc, img, invn);
  sim_kernel<<<(Bn * Nn) / 64, 128, 0, stream>>>(img, invn, pmask, prune);
  rel_kernel<<<(Bn * Nn) / 8, 256, 0, stream>>>(img, temb, pmask, prune, relm, nkept);
  select_kernel<<<Bn, 1024, 0, stream>>>(relm, nkept, idx);
  pair_kernel<<<dim3(Mm, Bn), 128, 0, stream>>>(img, idx, feat3);

  kv_kernel<<<dim3(Mm / 16, Bn), 256, 0, stream>>>(feat3, Wkt, bk, kbuf);
  kv_kernel<<<dim3(Mm / 16, Bn), 256, 0, stream>>>(feat3, Wvt, bv, vbuf);
  q_kernel<<<Bn, 256, 0, stream>>>(temb, Wq, bq, qbuf);
  attn_kernel<<<dim3(Hh, Bn), 256, 0, stream>>>(qbuf, kbuf, vbuf, ctx);
  out_kernel<<<Bn, 256, 0, stream>>>(ctx, Wo, bo, Wcls, bcls, logits);

  (void)in_sizes; (void)n_in; (void)out_size;
}
